// Categorical2DSemanticMapModule_91250875171564
// MI455X (gfx1250) — matrix operand order repaired and checked by reading, not compiled, not run on hardware
//
#include <hip/hip_runtime.h>
#include <cmath>

// ---------------- problem constants (from the reference) ----------------
#define Hs        480
#define Ws        640
#define DU        4
#define hD        (Hs / DU)          // 120
#define wD        (Ws / DU)          // 160
#define NP        (hD * wD)          // 19200 pixels per frame (downsampled)
#define NSEM      16
#define NCH_IN    (4 + NSEM)         // 20 input channels
#define NCH_OUT   (2 + NSEM)         // 18 output channels
#define VR        100
#define NCELL     (VR * VR)          // 10000 map cells
#define ZBINS     80
#define MIN_MAPPED 13                // agent-height z range [13,25)
#define MAX_MAPPED 25
#define NFRM      16                 // B*T
#define TPB       1024               // 32 wave32 per workgroup
#define NBLK      ((NP + TPB - 1) / TPB)   // 19 pixel blocks

// padded matrix geometry for the WMMA dilation: 100 -> 112 (7x16) M/N, 128 (4x32) K
#define MT        112
#define KT        128
#define NTIL      7                  // 112/16 tiles per dimension

// LDS pool layout (bytes):
//   [0,      40000) : u32/f32 histogram  -- reused as f16 T (M*B) after binarize
//   [40960,  69632) : f16 Ocm (column-major binary obstacle) -- sem tasks alias
//                     this region for the 8KB async staging buffers
#define OFF_OCM   40960
#define POOL_BYTES (OFF_OCM + MT * KT * 2)   // 69632

#if defined(__gfx1250__)
#define ASYNC_OK 1
#else
#define ASYNC_OK 0
#endif

typedef __attribute__((ext_vector_type(16))) _Float16 v16h;
typedef __attribute__((ext_vector_type(8)))  float    v8f;

// tridiagonal "ones band" matrix entry: M[i][j] = (|i-j|<=1), zero outside the
// live 100x100 region (this is what reproduces 'SAME' zero padding and also
// annihilates any K-padding garbage).
__device__ __forceinline__ _Float16 bandv(int i, int j)
{
    unsigned d = (unsigned)(i - j + 1);
    return (d <= 2u && i < VR && j < VR) ? (_Float16)1.0f : (_Float16)0.0f;
}

// Back-project one downsampled pixel -> map cell (or -1 if invalid).
__device__ __forceinline__ int bin_pixel(float d, int u, int r, float inv_f, bool* agent)
{
    float t  = d * inv_f;                                  // depth / F_PIX
    float X  = ((float)(u * DU) - 320.0f) * t;             // (uu - CX) * depth / F
    float Zh = 88.0f + (240.0f - (float)(r * DU)) * t;     // AGENT_H + (CY - vv) * depth / F
    int xb = (int)rintf(X / 5.0f + 50.0f);                 // round(X/XY_RES + VR/2)
    int yb = (int)rintf(d / 5.0f);                         // round(depth/XY_RES)
    int zb = (int)rintf(Zh / 5.0f) + 8;                    // round(Zh/Z_RES) - MIN_VOX
    bool valid = (d > 20.0f) & (d < 500.0f) &
                 (xb >= 0) & (xb < VR) &
                 (yb >= 0) & (yb < VR) &
                 (zb >= 0) & (zb < ZBINS);
    *agent = (zb >= MIN_MAPPED) & (zb < MAX_MAPPED);
    return valid ? (yb * VR + xb) : -1;
}

// grid = (NFRM frames, 1 + NSEM tasks), 1024 threads (32 wave32).
// task 0     : occupancy histogram -> explored + WMMA-dilated obstacle
// task 1..16 : one semantic channel histogram -> clip(sum/CAT_T)
__global__ __launch_bounds__(TPB)
void semmap_kernel(const float* __restrict__ obs, float* __restrict__ out, float inv_f)
{
    const int frame = blockIdx.x;
    const int task  = blockIdx.y;
    const int tid   = threadIdx.x;

    const float* __restrict__ depth =
        obs + ((size_t)frame * NCH_IN + 3) * (size_t)(Hs * Ws);

    __shared__ __align__(16) unsigned char pool[POOL_BYTES];
    unsigned int* accU = (unsigned int*)pool;          // packed occupancy counts
    float*        accF = (float*)pool;                 // sem-channel histogram
    _Float16*     Tm   = (_Float16*)pool;              // T = M*B (aliases accU)
    _Float16*     Ocm  = (_Float16*)(pool + OFF_OCM);  // binary obstacle, column-major
#if ASYNC_OK
    float (*stage)[TPB] = (float(*)[TPB])(pool + OFF_OCM);  // sem task staging
#endif

    for (int c = tid; c < NCELL; c += TPB) accU[c] = 0u;
    __syncthreads();

    if (task == 0) {
        // ---- occupancy: lo16 = any-z count (explored), hi16 = agent-z count ----
        for (int p = tid; p < NP; p += TPB) {
            int pn = p + TPB;                      // prefetch next strided depth sample
            if (pn < NP) {
                int rn = pn / wD, un = pn - rn * wD;
                __builtin_prefetch(&depth[(size_t)rn * DU * Ws + (size_t)un * DU], 0, 1);
            }
            int r = p / wD, u = p - r * wD;
            float d = depth[(size_t)r * DU * Ws + (size_t)u * DU];
            bool agent; int cell = bin_pixel(d, u, r, inv_f, &agent);
            if (cell >= 0)
                atomicAdd(&accU[cell], 1u | (agent ? 0x10000u : 0u));   // ds_add_u32
        }
        __syncthreads();

        // explored = (count>0); build padded column-major binary obstacle:
        // Ocm[n=x][k=y] = obst[y][x], zero outside the 100x100 live region.
        float* __restrict__ outExp = out + ((size_t)frame * NCH_OUT + 1) * (size_t)NCELL;
        for (int c = tid; c < MT * KT; c += TPB) {
            int x = c >> 7;            // 0..111  (column index n)
            int y = c & (KT - 1);      // 0..127  (row index k)
            _Float16 v = (_Float16)0.0f;
            if (x < VR && y < VR) {
                unsigned cnt = accU[y * VR + x];
                outExp[y * VR + x] = (cnt & 0xFFFFu) ? 1.0f : 0.0f;
                v = (cnt >> 16) ? (_Float16)1.0f : (_Float16)0.0f;
            }
            Ocm[c] = v;
        }
        __syncthreads();

        // T region aliases the histogram; clear all padded K columns.
        for (int c = tid; c < MT * KT; c += TPB) Tm[c] = (_Float16)0.0f;
        __syncthreads();

        // ---- dilation as matrix algebra on the WMMA unit ----
        // dilated = ((M . B . M) > 0), M = tridiagonal ones band (exact in f16).
        const int lane = tid & 31;
        const int wv   = tid >> 5;
        const int half = lane >> 4;          // 0: lanes 0-15, 1: lanes 16-31
        const int l15  = lane & 15;

        // pass 1: T = M . B   (A = band synthesized, B = Ocm contiguous LDS reads)
        for (int tile = wv; tile < NTIL * NTIL; tile += TPB / 32) {
            int Mb = (tile / NTIL) * 16, Nb = (tile % NTIL) * 16;
            v8f c = {};
#pragma unroll
            for (int kc = 0; kc < 4; ++kc) {
                int Kb = kc * 32;
                v16h a, b;
                int i  = Mb + l15;           // A row
                int ka = Kb + half * 16;     // A: elem e <-> K = ka + e
#pragma unroll
                for (int e = 0; e < 16; ++e)
                    a[e] = bandv(i, ka + e);
                const _Float16* bp = &Ocm[(Nb + l15) * KT + Kb + half * 16];
#pragma unroll
                for (int e = 0; e < 16; ++e) b[e] = bp[e];   // B: elem e <-> K = Kb+16*half+e
                c = __builtin_amdgcn_wmma_f32_16x16x32_f16(false, a, false, b,
                                                           (short)0, c, false, false);
            }
#pragma unroll
            for (int r = 0; r < 8; ++r)      // C vgpr r: M = Mb + r + 8*half, N = Nb + l15
                Tm[(Mb + r + half * 8) * KT + (Nb + l15)] = (_Float16)c[r];
        }
        __syncthreads();

        // pass 2: S = T . M ; threshold + store (A = Tm contiguous reads, B = band)
        float* __restrict__ outObs = out + ((size_t)frame * NCH_OUT + 0) * (size_t)NCELL;
        for (int tile = wv; tile < NTIL * NTIL; tile += TPB / 32) {
            int Mb = (tile / NTIL) * 16, Nb = (tile % NTIL) * 16;
            v8f c = {};
#pragma unroll
            for (int kc = 0; kc < 4; ++kc) {
                int Kb = kc * 32;
                v16h a, b;
                const _Float16* ap = &Tm[(Mb + l15) * KT + Kb + half * 8];
#pragma unroll
                for (int e = 0; e < 8; ++e) { a[e] = ap[e]; a[e + 8] = ap[e + 16]; }
                int kb = Kb + half * 8;
#pragma unroll
                for (int e = 0; e < 16; ++e) b[e] = bandv(kb + e + ((e & 8) ? 8 : 0), Nb + l15);
                c = __builtin_amdgcn_wmma_f32_16x16x32_f16(false, a, false, b,
                                                           (short)0, c, false, false);
            }
#pragma unroll
            for (int r = 0; r < 8; ++r) {
                int M = Mb + r + half * 8, N = Nb + l15;
                if (M < VR && N < VR)
                    outObs[M * VR + N] = (c[r] > 0.0f) ? 1.0f : 0.0f;
            }
        }
    } else {
        // ---- one semantic channel: float histogram of sem values ----
        const int s = task - 1;
        const float* __restrict__ semp =
            obs + ((size_t)frame * NCH_IN + 4 + s) * (size_t)(Hs * Ws);

#if ASYNC_OK
        // Double-buffered async global->LDS gather of the strided sem samples.
        // EXEC-uniform issues keep ASYNCcnt uniform per wave; each lane consumes
        // only its own stage slot, so s_wait_asynccnt (no barrier) suffices.
        {
            int pc = (tid < NP) ? tid : NP - 1;
            int r = pc / wD, u = pc - r * wD;
            unsigned lds = (unsigned)(size_t)(void*)&stage[0][tid];
            unsigned long long ga =
                (unsigned long long)(size_t)(const void*)(semp + (size_t)r * DU * Ws + (size_t)u * DU);
            asm volatile("global_load_async_to_lds_b32 %0, %1, off" :: "v"(lds), "v"(ga) : "memory");
        }
        int buf = 0;
        for (int blk = 0; blk < NBLK; ++blk) {
            int p = blk * TPB + tid;
            if (blk + 1 < NBLK) {                          // issue next block's DMA
                int pn = p + TPB;
                int pc = (pn < NP) ? pn : NP - 1;
                int r = pc / wD, u = pc - r * wD;
                unsigned lds = (unsigned)(size_t)(void*)&stage[buf ^ 1][tid];
                unsigned long long ga =
                    (unsigned long long)(size_t)(const void*)(semp + (size_t)r * DU * Ws + (size_t)u * DU);
                asm volatile("global_load_async_to_lds_b32 %0, %1, off" :: "v"(lds), "v"(ga) : "memory");
            }
            // overlap the DMA with the depth load + binning VALU work
            int cell = -1; bool agent;
            if (p < NP) {
                int r = p / wD, u = p - r * wD;
                float d = depth[(size_t)r * DU * Ws + (size_t)u * DU];
                cell = bin_pixel(d, u, r, inv_f, &agent);
            }
            if (blk + 1 < NBLK) asm volatile("s_wait_asynccnt 0x1" ::: "memory");
            else                asm volatile("s_wait_asynccnt 0x0" ::: "memory");
            float v = stage[buf][tid];
            if (cell >= 0) atomicAdd(&accF[cell], v);       // ds_add_f32
            buf ^= 1;
        }
#else
        for (int p = tid; p < NP; p += TPB) {
            int r = p / wD, u = p - r * wD;
            float d = depth[(size_t)r * DU * Ws + (size_t)u * DU];
            bool agent; int cell = bin_pixel(d, u, r, inv_f, &agent);
            if (cell >= 0) {
                float v = semp[(size_t)r * DU * Ws + (size_t)u * DU];
                atomicAdd(&accF[cell], v);
            }
        }
#endif
        __syncthreads();

        // semmap = clip(sum / CAT_T, 0, 1); sums are >= 0
        float* __restrict__ outSem =
            out + ((size_t)frame * NCH_OUT + 2 + s) * (size_t)NCELL;
        for (int c = tid; c < NCELL; c += TPB) {
            float v = accF[c] / 5.0f;
            outSem[c] = (v > 1.0f) ? 1.0f : v;
        }
    }
}

extern "C" void kernel_launch(void* const* d_in, const int* in_sizes, int n_in,
                              void* d_out, int out_size, void* d_ws, size_t ws_size,
                              hipStream_t stream)
{
    (void)in_sizes; (void)n_in; (void)d_ws; (void)ws_size; (void)out_size;
    const float* obs = (const float*)d_in[0];   // seq_obs (B,T,20,480,640) f32
    float* out = (float*)d_out;                 // (B,T,18,100,100) f32

    // F_PIX exactly as numpy computes it (float64), then demoted like JAX does
    const double f_pix = (double)Ws / 2.0 / tan((79.0 / 2.0) * 3.14159265358979323846 / 180.0);
    const float inv_f = (float)(1.0 / f_pix);

    dim3 grid(NFRM, 1 + NSEM);                  // 16 frames x 17 channel tasks
    semmap_kernel<<<grid, TPB, 0, stream>>>(obs, out, inv_f);
}
